// BatchFFN_73126113181994
// MI455X (gfx1250) — matrix operand order repaired and checked by reading, not compiled, not run on hardware
//
#include <hip/hip_runtime.h>
#include <hip/hip_bf16.h>
#include <stdint.h>

typedef __attribute__((ext_vector_type(16))) __bf16 v16bf;
typedef __attribute__((ext_vector_type(8)))  __bf16 v8bf;
typedef __attribute__((ext_vector_type(8)))  float  v8f;

#define E_   8
#define T_   512
#define DIN  7168
#define DINT 2048

#define BM 128          // token tile
#define BN 128          // output-feature tile
#define BK 32           // K slab (bf16 WMMA depth)
#define LDSROW 40       // shorts per LDS row: 32 data + 8 pad (80B, keeps 16B alignment)

union Frag { v16bf v; uint4 q[2]; };

// A fragment (16x32 bf16): lane<16 -> K {0..7,16..23}; lane>=16 -> K {8..15,24..31}
__device__ __forceinline__ v16bf load_frag_A(const unsigned short* lds, int row, int lane) {
    const int k0 = (lane >> 4) * 8;
    Frag f;
    f.q[0] = *(const uint4*)(lds + row * LDSROW + k0);
    f.q[1] = *(const uint4*)(lds + row * LDSROW + k0 + 16);
    return f.v;
}

// B fragment (32x16 bf16): lane<16 -> K {0..7,16..23}; lane>=16 -> K {8..15,24..31}
__device__ __forceinline__ v16bf load_frag_B(const unsigned short* lds, int col, int lane) {
    const int ks = (lane >> 4) * 8;
    Frag f;
    f.q[0] = *(const uint4*)(lds + col * LDSROW + ks);
    f.q[1] = *(const uint4*)(lds + col * LDSROW + ks + 16);
    return f.v;
}

// 8x fp32 -> 8x bf16 (RNE) via vector fptrunc: lowers to packed v_cvt_pk_bf16_f32
__device__ __forceinline__ void cvt_store8(unsigned short* dst, float4 a, float4 b) {
    v8f f = {a.x, a.y, a.z, a.w, b.x, b.y, b.z, b.w};
    union { v8bf v; uint4 q; } u;
    u.v = __builtin_convertvector(f, v8bf);
    *(uint4*)dst = u.q;
}

__device__ __forceinline__ unsigned short bf_bits(float f) {
    union { __bf16 b; unsigned short u; } cv;
    cv.b = (__bf16)f;
    return cv.u;
}

__device__ __forceinline__ v8f wmma_bf16(v16bf a, v16bf b, v8f c) {
    return __builtin_amdgcn_wmma_f32_16x16x32_bf16(false, a, false, b, (short)0, c, false, false);
}

// async copy of one 16B chunk into LDS (tracked by ASYNCcnt, no VGPR data)
__device__ __forceinline__ void async_cp16(unsigned short* lds_dst, const unsigned short* gsrc) {
    const unsigned int       ldst = (unsigned int)(uintptr_t)lds_dst;
    const unsigned long long gadr = (unsigned long long)(uintptr_t)gsrc;
    asm volatile("global_load_async_to_lds_b128 %0, %1, off"
                 :: "v"(ldst), "v"(gadr) : "memory");
}
__device__ __forceinline__ void wait_async0() {
    asm volatile("s_wait_asynccnt 0x0" ::: "memory");
}

// ============================ Kernel 1 helpers ============================
__device__ __forceinline__ void loadG1(const float* xe, const float* w1e, const float* w3e,
                                       int k, int tid, float4 r[12]) {
    #pragma unroll
    for (int i = 0; i < 2; ++i) {
        const int idx = tid + i * 256;
        const int row = idx >> 2;
        const int c   = (idx & 3) * 8;
        const size_t off = (size_t)row * DIN + k + c;
        const float4* px = (const float4*)(xe  + off);
        const float4* p1 = (const float4*)(w1e + off);
        const float4* p3 = (const float4*)(w3e + off);
        r[i*6+0] = px[0]; r[i*6+1] = px[1];
        r[i*6+2] = p1[0]; r[i*6+3] = p1[1];
        r[i*6+4] = p3[0]; r[i*6+5] = p3[1];
    }
}

__device__ __forceinline__ void storeL1(unsigned short* lx, unsigned short* l1s,
                                        unsigned short* l3s, int tid, const float4 r[12]) {
    #pragma unroll
    for (int i = 0; i < 2; ++i) {
        const int idx = tid + i * 256;
        const int row = idx >> 2;
        const int c   = (idx & 3) * 8;
        cvt_store8(&lx [row * LDSROW + c], r[i*6+0], r[i*6+1]);
        cvt_store8(&l1s[row * LDSROW + c], r[i*6+2], r[i*6+3]);
        cvt_store8(&l3s[row * LDSROW + c], r[i*6+4], r[i*6+5]);
    }
}

__device__ __forceinline__ void compute1(const unsigned short* lx, const unsigned short* l1s,
                                         const unsigned short* l3s, int wv, int lane,
                                         v8f acc1[8], v8f acc3[8]) {
    const v16bf fB1 = load_frag_B(l1s, wv * 16 + (lane & 15), lane);
    const v16bf fB3 = load_frag_B(l3s, wv * 16 + (lane & 15), lane);
    #pragma unroll
    for (int mt = 0; mt < 8; ++mt) {
        const v16bf fA = load_frag_A(lx, mt * 16 + (lane & 15), lane);
        acc1[mt] = wmma_bf16(fA, fB1, acc1[mt]);
        acc3[mt] = wmma_bf16(fA, fB3, acc3[mt]);
    }
}

// ---------------- Kernel 1: g = silu(x@w1^T + b1) * (x@w3^T + b3)  [bf16 out] --------------
__global__ __launch_bounds__(256)
void ffn_gate_kernel(const float* __restrict__ x,  const float* __restrict__ w1,
                     const float* __restrict__ b1, const float* __restrict__ w3,
                     const float* __restrict__ b3, unsigned short* __restrict__ g)
{
    __shared__ __align__(16) unsigned short lxA[BM * LDSROW], lxB[BM * LDSROW];
    __shared__ __align__(16) unsigned short l1A[BN * LDSROW], l1B[BN * LDSROW];
    __shared__ __align__(16) unsigned short l3A[BN * LDSROW], l3B[BN * LDSROW];

    const int e  = blockIdx.z;
    const int t0 = blockIdx.y * BM;
    const int f0 = blockIdx.x * BN;

    const int tid  = threadIdx.x;
    const int lane = tid & 31;
    const int wv   = tid >> 5;            // 0..7, owns F-columns [wv*16, wv*16+16)

    const float* xe  = x  + (size_t)e * T_   * DIN + (size_t)t0 * DIN;
    const float* w1e = w1 + (size_t)e * DINT * DIN + (size_t)f0 * DIN;
    const float* w3e = w3 + (size_t)e * DINT * DIN + (size_t)f0 * DIN;

    v8f acc1[8] = {};
    v8f acc3[8] = {};

    float4 r[12];
    loadG1(xe, w1e, w3e, 0, tid, r);
    storeL1(lxA, l1A, l3A, tid, r);
    loadG1(xe, w1e, w3e, BK, tid, r);
    __syncthreads();

    for (int k = 0; k < DIN; k += 2 * BK) {
        compute1(lxA, l1A, l3A, wv, lane, acc1, acc3);
        storeL1(lxB, l1B, l3B, tid, r);
        { int kn = k + 2 * BK; if (kn >= DIN) kn = 0; loadG1(xe, w1e, w3e, kn, tid, r); }
        __syncthreads();

        compute1(lxB, l1B, l3B, wv, lane, acc1, acc3);
        storeL1(lxA, l1A, l3A, tid, r);
        { int kn = k + 3 * BK; if (kn >= DIN) kn = 0; loadG1(xe, w1e, w3e, kn, tid, r); }
        __syncthreads();
    }

    // epilogue: bias + SwiGLU, write bf16 g[e, t, f]
    const int   fcol = f0 + wv * 16 + (lane & 15);
    const float bb1  = b1[(size_t)e * DINT + fcol];
    const float bb3  = b3[(size_t)e * DINT + fcol];
    unsigned short* ge = g + (size_t)e * T_ * DINT;
    #pragma unroll
    for (int mt = 0; mt < 8; ++mt) {
        #pragma unroll
        for (int rr = 0; rr < 8; ++rr) {
            const int   trow = t0 + mt * 16 + rr + ((lane >> 4) << 3);
            const float h1 = acc1[mt][rr] + bb1;
            const float h3 = acc3[mt][rr] + bb3;
            const float s  = h1 / (1.0f + __expf(-h1));   // silu
            ge[(size_t)trow * DINT + fcol] = bf_bits(s * h3);
        }
    }
}

// ============================ Kernel 2 helpers ============================
__device__ __forceinline__ void loadGw(const float* w2e, int k, int tid, float4 r[4]) {
    #pragma unroll
    for (int i = 0; i < 2; ++i) {
        const int idx = tid + i * 256;
        const int row = idx >> 2;
        const int c   = (idx & 3) * 8;
        const float4* pw = (const float4*)(w2e + (size_t)row * DINT + k + c);
        r[i*2+0] = pw[0]; r[i*2+1] = pw[1];
    }
}

__device__ __forceinline__ void storeLw(unsigned short* lw, int tid, const float4 r[4]) {
    #pragma unroll
    for (int i = 0; i < 2; ++i) {
        const int idx = tid + i * 256;
        const int row = idx >> 2;
        const int c   = (idx & 3) * 8;
        cvt_store8(&lw[row * LDSROW + c], r[i*2+0], r[i*2+1]);
    }
}

// g tile is already bf16: stage via async DMA to LDS (ASYNCcnt path)
__device__ __forceinline__ void asyncG(unsigned short* lg, const unsigned short* ge,
                                       int k, int tid) {
    #pragma unroll
    for (int i = 0; i < 2; ++i) {
        const int idx = tid + i * 256;
        const int row = idx >> 2;
        const int c   = (idx & 3) * 8;
        async_cp16(lg + row * LDSROW + c, ge + (size_t)row * DINT + k + c);
    }
}

__device__ __forceinline__ void compute2(const unsigned short* lg, const unsigned short* lw,
                                         int wv, int lane, v8f acc[8]) {
    const v16bf fB = load_frag_B(lw, wv * 16 + (lane & 15), lane);
    #pragma unroll
    for (int mt = 0; mt < 8; ++mt) {
        const v16bf fA = load_frag_A(lg, mt * 16 + (lane & 15), lane);
        acc[mt] = wmma_bf16(fA, fB, acc[mt]);
    }
}

// ---------------- Kernel 2: out = g @ w2^T + b2  [fp32 out] --------------------------------
__global__ __launch_bounds__(256)
void ffn_out_kernel(const unsigned short* __restrict__ g, const float* __restrict__ w2,
                    const float* __restrict__ b2, float* __restrict__ out)
{
    __shared__ __align__(16) unsigned short lgA[BM * LDSROW], lgB[BM * LDSROW];
    __shared__ __align__(16) unsigned short lwA[BN * LDSROW], lwB[BN * LDSROW];

    const int e  = blockIdx.z;
    const int t0 = blockIdx.y * BM;
    const int d0 = blockIdx.x * BN;

    const int tid  = threadIdx.x;
    const int lane = tid & 31;
    const int wv   = tid >> 5;

    const unsigned short* ge = g  + (size_t)e * T_  * DINT + (size_t)t0 * DINT;
    const float*         w2e = w2 + (size_t)e * DIN * DINT + (size_t)d0 * DINT;

    v8f acc[8] = {};

    float4 rw[4];
    loadGw(w2e, 0, tid, rw);
    storeLw(lwA, tid, rw);
    loadGw(w2e, BK, tid, rw);
    asyncG(lgA, ge, 0, tid);
    wait_async0();
    __syncthreads();

    for (int k = 0; k < DINT; k += 2 * BK) {
        asyncG(lgB, ge, k + BK, tid);                 // k+BK < DINT always (2048 = 64*32)
        compute2(lgA, lwA, wv, lane, acc);
        storeLw(lwB, tid, rw);
        { int kn = k + 2 * BK; if (kn >= DINT) kn = 0; loadGw(w2e, kn, tid, rw); }
        wait_async0();
        __syncthreads();

        { int kn = k + 2 * BK; if (kn >= DINT) kn = 0; asyncG(lgA, ge, kn, tid); }
        compute2(lgB, lwB, wv, lane, acc);
        storeLw(lwA, tid, rw);
        { int kn = k + 3 * BK; if (kn >= DINT) kn = 0; loadGw(w2e, kn, tid, rw); }
        wait_async0();
        __syncthreads();
    }

    const int   dcol = d0 + wv * 16 + (lane & 15);
    const float bb   = b2[(size_t)e * DIN + dcol];
    #pragma unroll
    for (int mt = 0; mt < 8; ++mt) {
        #pragma unroll
        for (int rr = 0; rr < 8; ++rr) {
            const int trow = t0 + mt * 16 + rr + ((lane >> 4) << 3);
            out[((size_t)e * T_ + trow) * DIN + dcol] = acc[mt][rr] + bb;
        }
    }
}

extern "C" void kernel_launch(void* const* d_in, const int* in_sizes, int n_in,
                              void* d_out, int out_size, void* d_ws, size_t ws_size,
                              hipStream_t stream) {
    (void)in_sizes; (void)n_in; (void)out_size; (void)ws_size;
    const float* x  = (const float*)d_in[0];
    const float* w1 = (const float*)d_in[1];
    const float* b1 = (const float*)d_in[2];
    const float* w3 = (const float*)d_in[3];
    const float* b3 = (const float*)d_in[4];
    const float* w2 = (const float*)d_in[5];
    const float* b2 = (const float*)d_in[6];
    float* out = (float*)d_out;
    unsigned short* g = (unsigned short*)d_ws;   // bf16 intermediate, 8*512*2048*2 = 16 MiB

    dim3 grid1(DINT / BN, T_ / BM, E_);   // (16, 4, 8)
    ffn_gate_kernel<<<grid1, 256, 0, stream>>>(x, w1, b1, w3, b3, g);

    dim3 grid2(DIN / BN, T_ / BM, E_);    // (56, 4, 8)
    ffn_out_kernel<<<grid2, 256, 0, stream>>>(g, w2, b2, out);
}
